// GPT_38654705664229
// MI455X (gfx1250) — matrix operand order repaired and checked by reading, not compiled, not run on hardware
//
#include <hip/hip_runtime.h>

// ---------------------------------------------------------------------------
// Types for CDNA5 WMMA
// ---------------------------------------------------------------------------
typedef __attribute__((ext_vector_type(16))) __bf16 v16bf;
typedef __attribute__((ext_vector_type(8)))  float  v8f;

union Frag  { unsigned int u[8]; v16bf v; };
union U16x8 { uint4 v; unsigned short s[8]; };

struct Stage { uint4 a0, a1, b0, b1; };

__device__ __forceinline__ unsigned short f32_to_bf16(float f) {
    unsigned int u = __float_as_uint(f);
    u += 0x7FFFu + ((u >> 16) & 1u);   // round-to-nearest-even
    return (unsigned short)(u >> 16);
}

__device__ __forceinline__ float gelu_tanh(float x) {
    float x3 = x * x * x;
    float t  = tanhf(0.7978845608028654f * (x + 0.044715f * x3));
    return 0.5f * x * (1.0f + t);
}

// ---------------------------------------------------------------------------
// bf16 GEMM:  Out[m,n] = epilogue( scale * sum_k A[m,k]*B[k,n] + bias[n] )
//   BT:       B element (k,n) comes from Bsrc[n*ldb + k]  (used for Q@K^T)
//   OUT_BF16: store bf16 (feeds next GEMM) vs fp32
//   EPI:      0 = none, 1 = tanh-GELU, 2 = += resid (fp32 residual stream)
//   NGUARD:   N not a multiple of the tile -> guard B loads / stores
//   HAS_BIAS: statically enable the bias add
// Branchless double-buffered LDS pipeline (nk >= 2 required; true here):
//   stage(0); barrier; { load(k+1); wmma(k); store(k+1); barrier } ; wmma(last)
// Assumes M % 128 == 0, K % 32 == 0. NbPad (multiple of 8) is the padded
// width of the B source, so a staged uint4 is fully in range or fully out.
// ---------------------------------------------------------------------------
#define BM 128
#define BN 128
#define BKK 32
#define LDA_S 40    // LDS row stride (ushorts) for A tile
#define LDB_S 136   // LDS row stride (ushorts) for B tile

template<bool BT, bool OUT_BF16, int EPI, bool NGUARD, bool HAS_BIAS>
__global__ __launch_bounds__(256, 1) void gemm_bf16_kernel(
    const unsigned short* __restrict__ A, int lda, long aOuter, long aInner,
    const unsigned short* __restrict__ Bsrc, int ldb, long bOuter, long bInner,
    const float* __restrict__ bias,
    const float* __restrict__ resid,
    void* __restrict__ Out, int ldc, long cOuter, long cInner,
    int M, int N, int NbPad, int K, int ZH, float scale)
{
    __shared__ unsigned short As[2][BM * LDA_S];
    __shared__ unsigned short Bs[2][BKK * LDB_S];

    const int tid  = threadIdx.x;
    const int lane = tid & 31;
    const int wave = tid >> 5;
    const int wm   = (wave & 3) * 32;   // wave tile M offset
    const int wn   = (wave >> 2) * 64;  // wave tile N offset

    const int z  = blockIdx.z;
    const int zb = z / ZH;
    const int zh = z - zb * ZH;
    A    += (long)zb * aOuter + (long)zh * aInner;
    Bsrc += (long)zb * bOuter + (long)zh * bInner;
    const long outBase = (long)zb * cOuter + (long)zh * cInner;

    const int m0 = blockIdx.y * BM;
    const int n0 = blockIdx.x * BN;

    // precomputed per-thread staging coordinates
    const int ar0 = (tid * 8) >> 5,          ac0 = (tid * 8) & 31;   // A pass 0
    const int ar1 = (tid * 8 + 2048) >> 5,   ac1 = (tid * 8 + 2048) & 31;
    const int br0 = (tid * 8) >> 7,          bc0 = (tid * 8) & 127;  // B pass 0
    const int br1 = (tid * 8 + 2048) >> 7,   bc1 = (tid * 8 + 2048) & 127;
    const int tn0 = tid >> 2,                tkg0 = (tid & 3) * 8;   // BT slots
    const int tn1 = (tid + 256) >> 2,        tkg1 = (tid & 3) * 8;

    auto loadTiles = [&](int k0) -> Stage {
        Stage st;
        st.a0 = *(const uint4*)(A + (long)(m0 + ar0) * lda + (k0 + ac0));
        st.a1 = *(const uint4*)(A + (long)(m0 + ar1) * lda + (k0 + ac1));
        if (!BT) {
            if (NGUARD && (n0 + bc0 + 8 > NbPad)) st.b0 = make_uint4(0u,0u,0u,0u);
            else st.b0 = *(const uint4*)(Bsrc + (long)(k0 + br0) * ldb + (n0 + bc0));
            if (NGUARD && (n0 + bc1 + 8 > NbPad)) st.b1 = make_uint4(0u,0u,0u,0u);
            else st.b1 = *(const uint4*)(Bsrc + (long)(k0 + br1) * ldb + (n0 + bc1));
        } else {
            if (NGUARD && (n0 + tn0 >= NbPad)) st.b0 = make_uint4(0u,0u,0u,0u);
            else st.b0 = *(const uint4*)(Bsrc + (long)(n0 + tn0) * ldb + (k0 + tkg0));
            if (NGUARD && (n0 + tn1 >= NbPad)) st.b1 = make_uint4(0u,0u,0u,0u);
            else st.b1 = *(const uint4*)(Bsrc + (long)(n0 + tn1) * ldb + (k0 + tkg1));
        }
        return st;
    };

    auto storeTiles = [&](int buf, Stage st) {
        *(uint4*)&As[buf][ar0 * LDA_S + ac0] = st.a0;
        *(uint4*)&As[buf][ar1 * LDA_S + ac1] = st.a1;
        if (!BT) {
            *(uint4*)&Bs[buf][br0 * LDB_S + bc0] = st.b0;
            *(uint4*)&Bs[buf][br1 * LDB_S + bc1] = st.b1;
        } else {
            U16x8 t0; t0.v = st.b0;
            U16x8 t1; t1.v = st.b1;
            #pragma unroll
            for (int i = 0; i < 8; ++i) {
                Bs[buf][(tkg0 + i) * LDB_S + tn0] = t0.s[i];
                Bs[buf][(tkg1 + i) * LDB_S + tn1] = t1.s[i];
            }
        }
    };

    // ---- accumulators ------------------------------------------------------
    v8f acc[2][4];
    #pragma unroll
    for (int i = 0; i < 2; ++i)
        #pragma unroll
        for (int j = 0; j < 4; ++j)
            #pragma unroll
            for (int r = 0; r < 8; ++r) acc[i][j][r] = 0.0f;

    auto computeTile = [&](int buf) {
        const unsigned short* Ab = &As[buf][0];
        const unsigned short* Bb = &Bs[buf][0];
        Frag afrag[2], bfrag[4];
        const int arow  = wm + (lane & 15);
        const int khalf = (lane >> 4) * 8;
        #pragma unroll
        for (int ms = 0; ms < 2; ++ms) {
            const unsigned short* ap = &Ab[(arow + ms * 16) * LDA_S];
            #pragma unroll
            for (int v = 0; v < 4; ++v)
                afrag[ms].u[v] = *(const unsigned int*)&ap[khalf + 2 * v];
            #pragma unroll
            for (int v = 4; v < 8; ++v)
                afrag[ms].u[v] = *(const unsigned int*)&ap[16 + khalf + 2 * (v - 4)];
        }
        #pragma unroll
        for (int ns = 0; ns < 4; ++ns) {
            const unsigned short* bp = &Bb[wn + ns * 16 + (lane & 15)];
            #pragma unroll
            for (int v = 0; v < 8; ++v) {
                const int kk = (v < 4) ? (khalf + 2 * v) : (16 + khalf + 2 * (v - 4));
                bfrag[ns].u[v] = (unsigned int)bp[kk * LDB_S] | ((unsigned int)bp[(kk + 1) * LDB_S] << 16);
            }
        }
        #pragma unroll
        for (int ms = 0; ms < 2; ++ms)
            #pragma unroll
            for (int ns = 0; ns < 4; ++ns)
                acc[ms][ns] = __builtin_amdgcn_wmma_f32_16x16x32_bf16(
                    false, afrag[ms].v, false, bfrag[ns].v,
                    (short)0, acc[ms][ns], false, false);
    };

    const int nk = K / BKK;   // >= 2 for every launch in this model

    storeTiles(0, loadTiles(0));
    __syncthreads();

    for (int kt = 0; kt < nk - 1; ++kt) {
        Stage st = loadTiles((kt + 1) * BKK);
        if (kt + 2 < nk) {
            // speculative prefetch of the tile after next (global_prefetch_b8)
            __builtin_prefetch(A + (long)(m0 + (tid >> 1)) * lda
                                 + (kt + 2) * BKK + (tid & 1) * 16, 0, 3);
        }
        computeTile(kt & 1);
        storeTiles((kt + 1) & 1, st);
        __syncthreads();
    }
    computeTile((nk - 1) & 1);

    // ---- epilogue + store (C/D layout: VGPR r -> M=r(+8), lane&15 -> N) ----
    const int rowHalf = (lane >> 4) * 8;
    const int colOff  = lane & 15;
    float*          outF = (float*)Out;
    unsigned short* outH = (unsigned short*)Out;
    #pragma unroll
    for (int ns = 0; ns < 4; ++ns) {
        const int col = n0 + wn + ns * 16 + colOff;
        const bool ok = (!NGUARD || col < N);
        float bv = 0.0f;
        if (HAS_BIAS && ok) bv = bias[col];
        #pragma unroll
        for (int ms = 0; ms < 2; ++ms)
            #pragma unroll
            for (int r = 0; r < 8; ++r) {
                if (ok) {
                    int row = m0 + wm + ms * 16 + rowHalf + r;
                    float v = acc[ms][ns][r] * scale + bv;
                    if (EPI == 1) v = gelu_tanh(v);
                    if (EPI == 2) v += resid[(long)row * ldc + col];
                    long offc = outBase + (long)row * ldc + col;
                    if (OUT_BF16) outH[offc] = f32_to_bf16(v);
                    else          outF[offc] = v;
                }
            }
    }
}

// ---------------------------------------------------------------------------
// fp32 -> bf16 cast with optional row padding (zero-fill) so every GEMM
// B-tile load is 16B aligned (lm_w: 50257 -> 50264 columns).
// ---------------------------------------------------------------------------
__global__ __launch_bounds__(256) void cast_pad_bf16(
    const float* __restrict__ src, unsigned short* __restrict__ dst,
    long rows, int cols, int ldDst)
{
    long n = rows * (long)ldDst;
    for (long i = blockIdx.x * 256L + threadIdx.x; i < n; i += gridDim.x * 256L) {
        long r = i / ldDst;
        int  c = (int)(i - r * ldDst);
        float v = (c < cols) ? src[r * (long)cols + c] : 0.0f;
        dst[i] = f32_to_bf16(v);
    }
}

// ---------------------------------------------------------------------------
// Embedding: x[i,:] = wte[idx[i],:] + wpe[i % T, :]
// ---------------------------------------------------------------------------
__global__ __launch_bounds__(256) void embed_kernel(
    const int* __restrict__ idx, const float* __restrict__ wte,
    const float* __restrict__ wpe, float* __restrict__ x, int Cd, int T)
{
    int i = blockIdx.x;
    int tok = idx[i];
    int t = i % T;
    for (int c = threadIdx.x; c < Cd; c += 256)
        x[(long)i * Cd + c] = wte[(long)tok * Cd + c] + wpe[(long)t * Cd + c];
}

// ---------------------------------------------------------------------------
// LayerNorm (fp32 in) -> bf16 out
// ---------------------------------------------------------------------------
__global__ __launch_bounds__(256) void layernorm_bf16(
    const float* __restrict__ x, const float* __restrict__ g,
    const float* __restrict__ bta, unsigned short* __restrict__ out, int Cd)
{
    __shared__ float sdata[256];
    const int tid = threadIdx.x;
    long row = blockIdx.x;
    const float* xr = x + row * Cd;

    float s = 0.0f;
    for (int c = tid; c < Cd; c += 256) s += xr[c];
    sdata[tid] = s; __syncthreads();
    for (int st = 128; st > 0; st >>= 1) {
        if (tid < st) sdata[tid] += sdata[tid + st];
        __syncthreads();
    }
    float mean = sdata[0] / (float)Cd;
    __syncthreads();

    float vs = 0.0f;
    for (int c = tid; c < Cd; c += 256) { float d = xr[c] - mean; vs += d * d; }
    sdata[tid] = vs; __syncthreads();
    for (int st = 128; st > 0; st >>= 1) {
        if (tid < st) sdata[tid] += sdata[tid + st];
        __syncthreads();
    }
    float rstd = rsqrtf(sdata[0] / (float)Cd + 1e-5f);

    for (int c = tid; c < Cd; c += 256)
        out[row * Cd + c] = f32_to_bf16((xr[c] - mean) * rstd * g[c] + bta[c]);
}

// ---------------------------------------------------------------------------
// Causal softmax: fp32 scores row -> bf16 probabilities (masked cols = 0)
// blockIdx.x = b*H*T + h*T + q
// ---------------------------------------------------------------------------
__global__ __launch_bounds__(256) void softmax_causal(
    const float* __restrict__ att, unsigned short* __restrict__ attbf, int T)
{
    __shared__ float sdata[256];
    const int tid = threadIdx.x;
    long row = blockIdx.x;
    int q = (int)(row % (long)T);
    const float* a = att + row * (long)T;
    unsigned short* ob = attbf + row * (long)T;

    float mx = -3.0e38f;
    for (int j = tid; j <= q; j += 256) mx = fmaxf(mx, a[j]);
    sdata[tid] = mx; __syncthreads();
    for (int st = 128; st > 0; st >>= 1) {
        if (tid < st) sdata[tid] = fmaxf(sdata[tid], sdata[tid + st]);
        __syncthreads();
    }
    mx = sdata[0]; __syncthreads();

    float sum = 0.0f;
    for (int j = tid; j <= q; j += 256) sum += __expf(a[j] - mx);
    sdata[tid] = sum; __syncthreads();
    for (int st = 128; st > 0; st >>= 1) {
        if (tid < st) sdata[tid] += sdata[tid + st];
        __syncthreads();
    }
    float inv = 1.0f / sdata[0];

    for (int j = tid; j < T; j += 256)
        ob[j] = (j <= q) ? f32_to_bf16(__expf(a[j] - mx) * inv) : (unsigned short)0;
}

// ---------------------------------------------------------------------------
// Host orchestration
// ---------------------------------------------------------------------------
extern "C" void kernel_launch(void* const* d_in, const int* in_sizes, int n_in,
                              void* d_out, int out_size, void* d_ws, size_t ws_size,
                              hipStream_t stream) {
    (void)in_sizes; (void)n_in; (void)out_size; (void)ws_size;

    const int*   idx     = (const int*)  d_in[0];
    const float* wte     = (const float*)d_in[1];
    const float* wpe     = (const float*)d_in[2];
    const float* ln1_g   = (const float*)d_in[3];
    const float* ln1_b   = (const float*)d_in[4];
    const float* attn_w  = (const float*)d_in[5];
    const float* attn_b  = (const float*)d_in[6];
    const float* proj_w  = (const float*)d_in[7];
    const float* proj_b  = (const float*)d_in[8];
    const float* ln2_g   = (const float*)d_in[9];
    const float* ln2_b   = (const float*)d_in[10];
    const float* fc_w    = (const float*)d_in[11];
    const float* fc_b    = (const float*)d_in[12];
    const float* cproj_w = (const float*)d_in[13];
    const float* cproj_b = (const float*)d_in[14];
    const float* lnf_g   = (const float*)d_in[15];
    const float* lnf_b   = (const float*)d_in[16];
    const float* lm_w    = (const float*)d_in[17];

    const int Lc = 12, H = 12, C = 768, V = 50257, T = 1024, Bq = 2;
    const int C3 = 3 * C, C4 = 4 * C, D = 64, M = Bq * T;
    const int Vpad = (V + 7) & ~7;   // 50264

    // ---- workspace carve-up (256B aligned) --------------------------------
    char* ws = (char*)d_ws;
    size_t off = 0;
    auto carve = [&](size_t bytes) -> char* {
        char* p = ws + off;
        off = (off + bytes + 255) & ~(size_t)255;
        return p;
    };
    unsigned short* attn_w_bf  = (unsigned short*)carve((size_t)Lc * C * C3 * 2);
    unsigned short* proj_w_bf  = (unsigned short*)carve((size_t)Lc * C * C  * 2);
    unsigned short* fc_w_bf    = (unsigned short*)carve((size_t)Lc * C * C4 * 2);
    unsigned short* cproj_w_bf = (unsigned short*)carve((size_t)Lc * C4 * C * 2);
    unsigned short* lm_w_bf    = (unsigned short*)carve((size_t)C * Vpad * 2);
    float*          xf         = (float*)         carve((size_t)M * C * 4);
    unsigned short* h_bf       = (unsigned short*)carve((size_t)M * C * 2);
    unsigned short* qkv_bf     = (unsigned short*)carve((size_t)M * C3 * 2);
    float*          att_f      = (float*)         carve((size_t)Bq * H * T * T * 4);
    unsigned short* att_bf     = (unsigned short*)carve((size_t)Bq * H * T * T * 2);
    unsigned short* y_bf       = (unsigned short*)carve((size_t)M * C * 2);
    unsigned short* m_bf       = (unsigned short*)carve((size_t)M * C4 * 2);

    // ---- cast weights to bf16 (lm_w padded to 8-aligned rows) -------------
    cast_pad_bf16<<<4096, 256, 0, stream>>>(attn_w,  attn_w_bf,  (long)Lc * C,  C3, C3);
    cast_pad_bf16<<<4096, 256, 0, stream>>>(proj_w,  proj_w_bf,  (long)Lc * C,  C,  C);
    cast_pad_bf16<<<4096, 256, 0, stream>>>(fc_w,    fc_w_bf,    (long)Lc * C,  C4, C4);
    cast_pad_bf16<<<4096, 256, 0, stream>>>(cproj_w, cproj_w_bf, (long)Lc * C4, C,  C);
    cast_pad_bf16<<<4096, 256, 0, stream>>>(lm_w,    lm_w_bf,    (long)C,       V,  Vpad);

    // ---- embedding --------------------------------------------------------
    embed_kernel<<<M, 256, 0, stream>>>(idx, wte, wpe, xf, C, T);

    // ---- transformer blocks ----------------------------------------------
    for (int l = 0; l < Lc; ++l) {
        layernorm_bf16<<<M, 256, 0, stream>>>(xf, ln1_g + (long)l * C,
                                              ln1_b + (long)l * C, h_bf, C);
        // qkv = h @ attn_w + attn_b            [M, 2304] bf16
        gemm_bf16_kernel<false, true, 0, false, true>
            <<<dim3(C3 / 128, M / 128, 1), 256, 0, stream>>>(
            h_bf, C, 0, 0,
            attn_w_bf + (long)l * C * C3, C3, 0, 0,
            attn_b + (long)l * C3, nullptr,
            qkv_bf, C3, 0, 0, M, C3, C3, C, 1, 1.0f);
        // scores = (1/8) q @ k^T  per (b,h)    [B,H,T,T] fp32
        gemm_bf16_kernel<true, false, 0, false, false>
            <<<dim3(T / 128, T / 128, Bq * H), 256, 0, stream>>>(
            qkv_bf, C3, (long)T * C3, D,
            qkv_bf + C, C3, (long)T * C3, D,
            nullptr, nullptr,
            att_f, T, (long)H * T * T, (long)T * T,
            T, T, T, D, H, 0.125f);
        softmax_causal<<<Bq * H * T, 256, 0, stream>>>(att_f, att_bf, T);
        // y = P @ v  per (b,h), scattered into [B,T,C] bf16
        gemm_bf16_kernel<false, true, 0, true, false>
            <<<dim3(1, T / 128, Bq * H), 256, 0, stream>>>(
            att_bf, T, (long)H * T * T, (long)T * T,
            qkv_bf + 2 * C, C3, (long)T * C3, D,
            nullptr, nullptr,
            y_bf, C, (long)T * C, D,
            T, D, D, T, H, 1.0f);
        // x = x + y @ proj_w + proj_b
        gemm_bf16_kernel<false, false, 2, false, true>
            <<<dim3(C / 128, M / 128, 1), 256, 0, stream>>>(
            y_bf, C, 0, 0,
            proj_w_bf + (long)l * C * C, C, 0, 0,
            proj_b + (long)l * C, xf,
            xf, C, 0, 0, M, C, C, C, 1, 1.0f);
        layernorm_bf16<<<M, 256, 0, stream>>>(xf, ln2_g + (long)l * C,
                                              ln2_b + (long)l * C, h_bf, C);
        // m = gelu(h2 @ fc_w + fc_b)           [M, 3072] bf16
        gemm_bf16_kernel<false, true, 1, false, true>
            <<<dim3(C4 / 128, M / 128, 1), 256, 0, stream>>>(
            h_bf, C, 0, 0,
            fc_w_bf + (long)l * C * C4, C4, 0, 0,
            fc_b + (long)l * C4, nullptr,
            m_bf, C4, 0, 0, M, C4, C4, C, 1, 1.0f);
        // x = x + m @ cproj_w + cproj_b
        gemm_bf16_kernel<false, false, 2, false, true>
            <<<dim3(C / 128, M / 128, 1), 256, 0, stream>>>(
            m_bf, C4, 0, 0,
            cproj_w_bf + (long)l * C4 * C, C, 0, 0,
            cproj_b + (long)l * C, xf,
            xf, C, 0, 0, M, C, C, C4, 1, 1.0f);
    }

    // ---- final LN + lm_head ----------------------------------------------
    layernorm_bf16<<<M, 256, 0, stream>>>(xf, lnf_g, lnf_b, h_bf, C);
    gemm_bf16_kernel<false, false, 0, true, false>
        <<<dim3((V + 127) / 128, M / 128, 1), 256, 0, stream>>>(
        h_bf, C, 0, 0,
        lm_w_bf, Vpad, 0, 0,
        nullptr, nullptr,
        (float*)d_out, V, 0, 0, M, V, Vpad, C, 1, 1.0f);
}
